// EncoderRNN_51092930953402
// MI455X (gfx1250) — matrix operand order repaired and checked by reading, not compiled, not run on hardware
//
#include <hip/hip_runtime.h>
#include <hip/hip_bf16.h>

// ---------------- problem dims ----------------
#define Ld   128
#define Bd   8
#define Dd   384
#define Hd   192
#define G4   768          // 4*H
#define NP   1024         // B*L rows in attention stage
#define KT_IH 12          // 384/32
#define KT_HH 6           // 192/32

typedef __attribute__((ext_vector_type(16))) __bf16 v16bf;
typedef __attribute__((ext_vector_type(8)))  float  v8f;

__device__ __forceinline__ v8f wmma_bf16(v16bf a, v16bf b, v8f c) {
  // D = A(16x32 bf16) x B(32x16 bf16) + C(16x16 f32)
  return __builtin_amdgcn_wmma_f32_16x16x32_bf16(false, a, false, b, (short)0, c,
                                                 false, false);
}

__device__ __forceinline__ float sigf(float x) { return 1.0f / (1.0f + __expf(-x)); }

// Build an A-fragment (16x32 bf16) from an f32 row-major matrix (used by one-shot GEMMs).
// A layout (wave32): lane l -> row m=l&15, half=l>>4; element e -> k = kbase + ((e&8)<<1) + half*8 + (e&7)
__device__ __forceinline__ v16bf make_a_frag(const float* base, int ld, int kbase, int mmax) {
  int lane = threadIdx.x & 31;
  int m    = lane & 15;
  int hl   = lane >> 4;
  v16bf a;
  if (m < mmax) {
    const float* rp = base + (size_t)m * ld + kbase + hl * 8;
    #pragma unroll
    for (int e = 0; e < 16; ++e) {
      int k = ((e & 8) << 1) + (e & 7);
      a[e] = (__bf16)rp[k];
    }
  } else {
    #pragma unroll
    for (int e = 0; e < 16; ++e) a[e] = (__bf16)0.0f;
  }
  return a;
}

// Load a pre-packed B fragment (32x16 bf16): one 32B vector load per lane.
__device__ __forceinline__ v16bf load_b_frag(const __bf16* pk, int frag) {
  int lane = threadIdx.x & 31;
  return *(const v16bf*)(pk + (((size_t)frag * 32 + lane) << 4));
}

// Scatter one h value (f32 -> bf16) into the LDS A-fragment-packed h buffer.
// Inverse of the A-layout mapping: for k within a 32-chunk:
//   hl = (k>>3)&1 ;  e = (k&7) | ((k>>4)&1)<<3 ;  lane position = hl*16 + m
__device__ __forceinline__ void store_h_pack(__bf16* pack, int m, int jc, float h2) {
  int kt = jc >> 5;
  int kk = jc & 31;
  int hl = (kk >> 3) & 1;
  int e  = (kk & 7) | (((kk >> 4) & 1) << 3);
  pack[(size_t)(kt * 32 + hl * 16 + m) * 16 + e] = (__bf16)h2;
}

// ---------------- weight packing ----------------
// W is [N=768, K] row-major f32 (gate-major). B-matrix for WMMA is W^T (KxN).
// Packed layout: frag f = ntile*KT + ktile; per-frag: lane(0..31) x 16 bf16;
// lane l: n = ntile*16 + (l&15); element e: k = ktile*32 + (l>>4)*8 + ((e&8)<<1) + (e&7).
__global__ void pack_w_kernel(const float* __restrict__ W, __bf16* __restrict__ out,
                              int K, int total) {
  int p = blockIdx.x * blockDim.x + threadIdx.x;
  if (p >= total) return;
  int KT   = K >> 5;
  int f    = p >> 9;          // /512 elems per frag
  int rem  = p & 511;
  int lane = rem >> 4;
  int e    = rem & 15;
  int nt = f / KT, kt = f % KT;
  int n = nt * 16 + (lane & 15);
  int k = kt * 32 + ((lane >> 4) << 3) + ((e & 8) << 1) + (e & 7);
  out[p] = (__bf16)W[(size_t)n * K + k];
}

// ---------------- generic GEMM: C[M,768] = A[M,K] * Wpk^T + bias ----------------
__global__ void gemm_xw_kernel(const float* __restrict__ A, const __bf16* __restrict__ Bpk,
                               const float* __restrict__ bias, float* __restrict__ C,
                               int K) {
  const int ntiles = G4 >> 4;              // 48
  int tile = blockIdx.x;
  int mt = tile / ntiles, nt = tile % ntiles;
  int KT = K >> 5;
  const float* Abase = A + (size_t)mt * 16 * K;
  v8f acc = {};
  for (int kt = 0; kt < KT; ++kt) {
    v16bf a = make_a_frag(Abase, K, kt * 32, 16);
    v16bf b = load_b_frag(Bpk, nt * KT + kt);
    acc = wmma_bf16(a, b, acc);
  }
  int lane = threadIdx.x & 31;
  int ncol = nt * 16 + (lane & 15);
  float bv = bias ? bias[ncol] : 0.0f;
  #pragma unroll
  for (int v = 0; v < 8; ++v) {
    int m = mt * 16 + v + ((lane >> 4) << 3);
    C[(size_t)m * G4 + ncol] = acc[v] + bv;
  }
}

// ---------------- phase 1: bidirectional masked LSTM (batch 8) ----------------
// blockIdx.x: 0=fwd, 1=bwd.  12 waves; wave w owns gate-column quad {w, w+12, w+24, w+36}.
// h lives in registers (owner lane) and, packed as bf16 A-fragments, in 6KB of LDS.
__global__ __launch_bounds__(384) void lstm_phase1_kernel(
    const float* __restrict__ XGF, const float* __restrict__ XGB,
    const __bf16* __restrict__ WHHF, const __bf16* __restrict__ WHHB,
    const int* __restrict__ lens,
    float* __restrict__ FWD, float* __restrict__ BWD) {
  __shared__ __align__(32) __bf16 hpk[KT_HH * 32 * 16];   // rows 8..15 stay zero
  int dir = blockIdx.x;
  const float*  XG = dir ? XGB : XGF;
  const __bf16* W  = dir ? WHHB : WHHF;
  float*        HS = dir ? BWD : FWD;
  int tid = threadIdx.x, wave = tid >> 5, lane = tid & 31;
  int half = lane >> 4;
  int jc   = wave * 16 + (lane & 15);      // hidden column 0..191
  for (int i = tid; i < KT_HH * 32 * 16; i += 384) hpk[i] = (__bf16)0.0f;
  float creg[8], hreg[8];
  #pragma unroll
  for (int v = 0; v < 8; ++v) { creg[v] = 0.0f; hreg[v] = 0.0f; }
  __syncthreads();

  for (int t = 0; t < Ld; ++t) {
    int tt = dir ? (Ld - 1 - t) : t;
    v16bf af[KT_HH];
    #pragma unroll
    for (int kt = 0; kt < KT_HH; ++kt)
      af[kt] = *(const v16bf*)(hpk + (size_t)(kt * 32 + lane) * 16);
    __syncthreads();                        // all A-frag reads done before h rewrites
    v8f acc[4];
    #pragma unroll
    for (int q = 0; q < 4; ++q) {
      v8f a = {};
      #pragma unroll
      for (int kt = 0; kt < KT_HH; ++kt)
        a = wmma_bf16(af[kt], load_b_frag(W, (q * 12 + wave) * KT_HH + kt), a);
      acc[q] = a;
    }
    if (half == 0) {                        // rows 0..7 = batch
      #pragma unroll
      for (int v = 0; v < 8; ++v) {
        int bb = v;
        const float* Xr = XG + (size_t)(tt * Bd + bb) * G4;
        float gi = acc[0][v] + Xr[jc];
        float gf = acc[1][v] + Xr[Hd + jc];
        float gg = acc[2][v] + Xr[2 * Hd + jc];
        float go = acc[3][v] + Xr[3 * Hd + jc];
        float cold = creg[v];
        float hold = hreg[v];
        float c2 = sigf(gf) * cold + sigf(gi) * tanhf(gg);
        float h2 = sigf(go) * tanhf(c2);
        if (tt >= lens[bb]) { c2 = cold; h2 = hold; }
        creg[v] = c2;
        hreg[v] = h2;
        store_h_pack(hpk, bb, jc, h2);
        HS[(size_t)(tt * Bd + bb) * Hd + jc] = h2;
      }
    }
    __syncthreads();                        // h writes visible before next step
  }
}

// ---------------- assemble bi / reversed contexts / output bi-part ----------------
__global__ void assemble_kernel(const float* __restrict__ FWD, const float* __restrict__ BWD,
                                const int* __restrict__ lens,
                                float* __restrict__ EMB, float* __restrict__ REV,
                                float* __restrict__ out) {
  int idx = blockIdx.x * blockDim.x + threadIdx.x;   // over 1024*384
  if (idx >= NP * (2 * Hd)) return;
  int j = idx % (2 * Hd);
  int row = idx / (2 * Hd);                // b*L + t
  int b = row >> 7, t = row & (Ld - 1);
  int len_b = lens[b];
  bool valid = t < len_b;
  float e = 0.0f;
  if (valid)
    e = (j < Hd) ? FWD[(size_t)(t * Bd + b) * Hd + j]
                 : BWD[(size_t)(t * Bd + b) * Hd + (j - Hd)];
  EMB[idx] = e;
  out[(size_t)((1 + t) * Bd + b) * G4 + j] = e;
  float rv = 0.0f;
  if (valid) {
    int ridx = len_b - 1 - t;              // in [0, len-1] when valid
    rv = (j < Hd) ? FWD[(size_t)(ridx * Bd + b) * Hd + j]
                  : BWD[(size_t)(ridx * Bd + b) * Hd + (j - Hd)];
  }
  REV[idx] = rv;
  if (t == 0) {                            // zero pad row 0 of the output
    out[(size_t)b * G4 + j] = 0.0f;
    out[(size_t)b * G4 + (2 * Hd) + j] = 0.0f;
  }
}

// ---------------- phase 2: 1024-row attention LSTMs (persistent per 16-row tile) ----
// grid = 128: side = blk>>6 (0=hL,1=hR), mtile = blk&63. Rows are independent -> no
// cross-workgroup sync; full 129-step scan inside the workgroup.
__global__ __launch_bounds__(384) void lstm_phase2_kernel(
    const float* __restrict__ XCL, const float* __restrict__ XCR,
    const float* __restrict__ XRR,
    const __bf16* __restrict__ WL, const __bf16* __restrict__ WR,
    const int* __restrict__ lens, float* __restrict__ out) {
  __shared__ __align__(32) __bf16 hpk[KT_HH * 32 * 16];
  int blk  = blockIdx.x;
  int side = blk >> 6;
  int mt   = blk & 63;
  int r0   = mt * 16;
  int b    = r0 / Ld;                      // 16 | 128 -> whole tile is one batch item
  int p0   = r0 % Ld;
  const __bf16* W    = side ? WR : WL;
  const float*  Xcur = side ? XCR : XCL;
  const float*  Xctx = side ? XRR : XCL;   // for hL, ctx rows == cur buffer rows
  int tid = threadIdx.x, wave = tid >> 5, lane = tid & 31;
  int half = lane >> 4;
  int jc   = wave * 16 + (lane & 15);
  int len_b = lens[b];
  for (int i = tid; i < KT_HH * 32 * 16; i += 384) hpk[i] = (__bf16)0.0f;
  float creg[8], hreg[8];
  #pragma unroll
  for (int v = 0; v < 8; ++v) { creg[v] = 0.0f; hreg[v] = 0.0f; }
  __syncthreads();

  for (int s = 0; s <= Ld; ++s) {          // 129 scan steps
    // prefetch next step's shared context row into the caches (gfx1250 global_prefetch)
    if (s < Ld) {
      const float* Pn = Xctx + (size_t)(b * Ld + s) * G4;
      __builtin_prefetch(Pn + jc, 0, 1);
      __builtin_prefetch(Pn + Hd + jc, 0, 1);
      __builtin_prefetch(Pn + 2 * Hd + jc, 0, 1);
      __builtin_prefetch(Pn + 3 * Hd + jc, 0, 1);
    }
    v16bf af[KT_HH];
    #pragma unroll
    for (int kt = 0; kt < KT_HH; ++kt)
      af[kt] = *(const v16bf*)(hpk + (size_t)(kt * 32 + lane) * 16);
    __syncthreads();
    v8f acc[4];
    #pragma unroll
    for (int q = 0; q < 4; ++q) {
      v8f a = {};
      #pragma unroll
      for (int kt = 0; kt < KT_HH; ++kt)
        a = wmma_bf16(af[kt], load_b_frag(W, (q * 12 + wave) * KT_HH + kt), a);
      acc[q] = a;
    }
    const float* X = (s == 0) ? Xcur : Xctx;
    size_t ctxrow  = (size_t)(b * Ld + (s > 0 ? s - 1 : 0));
    #pragma unroll
    for (int v = 0; v < 8; ++v) {
      int m   = v + half * 8;
      int pos = p0 + m;
      size_t xr = (s == 0) ? (size_t)(r0 + m) : ctxrow;
      const float* Xr = X + xr * G4;
      float gi = acc[0][v] + Xr[jc];
      float gf = acc[1][v] + Xr[Hd + jc];
      float gg = acc[2][v] + Xr[2 * Hd + jc];
      float go = acc[3][v] + Xr[3 * Hd + jc];
      int leng = ((pos < len_b) ? pos : 0) + 1;
      float cold = creg[v];
      float hold = hreg[v];
      float c2 = sigf(gf) * cold + sigf(gi) * tanhf(gg);
      float h2 = sigf(go) * tanhf(c2);
      if (s >= leng) { c2 = cold; h2 = hold; }
      creg[v] = c2;
      hreg[v] = h2;
      store_h_pack(hpk, m, jc, h2);
    }
    __syncthreads();
  }
  // store final hidden (exact f32 from registers) into att half of the output
  #pragma unroll
  for (int v = 0; v < 8; ++v) {
    int m = v + half * 8;
    int pos = p0 + m;
    out[(size_t)((1 + pos) * Bd + b) * G4 + (2 * Hd) + side * Hd + jc] = hreg[v];
  }
}

// ---------------- host launcher ----------------
extern "C" void kernel_launch(void* const* d_in, const int* in_sizes, int n_in,
                              void* d_out, int out_size, void* d_ws, size_t ws_size,
                              hipStream_t stream) {
  (void)in_sizes; (void)n_in; (void)out_size; (void)ws_size;
  const float* embs  = (const float*)d_in[0];   // [128,8,384]
  const int*   lens  = (const int*)  d_in[1];   // [8]
  const float* wihf  = (const float*)d_in[2];   // [768,384]
  const float* whhf  = (const float*)d_in[3];   // [768,192]
  const float* bf    = (const float*)d_in[4];
  const float* wihb  = (const float*)d_in[5];
  const float* whhb  = (const float*)d_in[6];
  const float* bb    = (const float*)d_in[7];
  const float* wlih  = (const float*)d_in[8];   // [768,384]
  const float* wlhh  = (const float*)d_in[9];
  const float* bl    = (const float*)d_in[10];
  const float* wrih  = (const float*)d_in[11];
  const float* wrhh  = (const float*)d_in[12];
  const float* br    = (const float*)d_in[13];
  float* out = (float*)d_out;

  char* ws = (char*)d_ws;
  size_t o = 0;
  auto take = [&](size_t n) { void* p = ws + o; o += (n + 255) & ~(size_t)255; return p; };
  const size_t IH_B = (size_t)G4 * Dd * 2;      // packed bf16 bytes [768,384]
  const size_t HH_B = (size_t)G4 * Hd * 2;      // packed bf16 bytes [768,192]
  __bf16* pk_wihf = (__bf16*)take(IH_B);
  __bf16* pk_wihb = (__bf16*)take(IH_B);
  __bf16* pk_whhf = (__bf16*)take(HH_B);
  __bf16* pk_whhb = (__bf16*)take(HH_B);
  __bf16* pk_wlih = (__bf16*)take(IH_B);
  __bf16* pk_wlhh = (__bf16*)take(HH_B);
  __bf16* pk_wrih = (__bf16*)take(IH_B);
  __bf16* pk_wrhh = (__bf16*)take(HH_B);
  float* XGF = (float*)take((size_t)NP * G4 * 4);
  float* XGB = (float*)take((size_t)NP * G4 * 4);
  float* FWD = (float*)take((size_t)Ld * Bd * Hd * 4);
  float* BWD = (float*)take((size_t)Ld * Bd * Hd * 4);
  float* EMB = (float*)take((size_t)NP * 2 * Hd * 4);
  float* REV = (float*)take((size_t)NP * 2 * Hd * 4);
  float* XCL = (float*)take((size_t)NP * G4 * 4);
  float* XCR = (float*)take((size_t)NP * G4 * 4);
  float* XRR = (float*)take((size_t)NP * G4 * 4);

  // 1) pack weights into WMMA B-fragment layout (bf16)
  const int ihTot = G4 * Dd, hhTot = G4 * Hd;
  dim3 pb(256);
  pack_w_kernel<<<(ihTot + 255) / 256, pb, 0, stream>>>(wihf, pk_wihf, Dd, ihTot);
  pack_w_kernel<<<(ihTot + 255) / 256, pb, 0, stream>>>(wihb, pk_wihb, Dd, ihTot);
  pack_w_kernel<<<(hhTot + 255) / 256, pb, 0, stream>>>(whhf, pk_whhf, Hd, hhTot);
  pack_w_kernel<<<(hhTot + 255) / 256, pb, 0, stream>>>(whhb, pk_whhb, Hd, hhTot);
  pack_w_kernel<<<(ihTot + 255) / 256, pb, 0, stream>>>(wlih, pk_wlih, Dd, ihTot);
  pack_w_kernel<<<(hhTot + 255) / 256, pb, 0, stream>>>(wlhh, pk_wlhh, Hd, hhTot);
  pack_w_kernel<<<(ihTot + 255) / 256, pb, 0, stream>>>(wrih, pk_wrih, Dd, ihTot);
  pack_w_kernel<<<(hhTot + 255) / 256, pb, 0, stream>>>(wrhh, pk_wrhh, Hd, hhTot);

  // 2) hoisted input GEMMs for phase-1 (x @ w_ih.T + b), rows = t*8+b
  const int gemmBlocks = (NP / 16) * (G4 / 16);   // 3072
  gemm_xw_kernel<<<gemmBlocks, 32, 0, stream>>>(embs, pk_wihf, bf, XGF, Dd);
  gemm_xw_kernel<<<gemmBlocks, 32, 0, stream>>>(embs, pk_wihb, bb, XGB, Dd);

  // 3) bidirectional masked LSTM (persistent, one workgroup per direction)
  lstm_phase1_kernel<<<2, 384, 0, stream>>>(XGF, XGB, pk_whhf, pk_whhb, lens, FWD, BWD);

  // 4) assemble bi embeddings, reversed contexts, output bi-part + zero row
  assemble_kernel<<<(NP * 2 * Hd + 255) / 256, 256, 0, stream>>>(FWD, BWD, lens, EMB, REV, out);

  // 5) hoisted input GEMMs for phase-2 (serve both cur and shared ctx rows)
  gemm_xw_kernel<<<gemmBlocks, 32, 0, stream>>>(EMB, pk_wlih, bl, XCL, 2 * Hd);
  gemm_xw_kernel<<<gemmBlocks, 32, 0, stream>>>(EMB, pk_wrih, br, XCR, 2 * Hd);
  gemm_xw_kernel<<<gemmBlocks, 32, 0, stream>>>(REV, pk_wrih, br, XRR, 2 * Hd);

  // 6) attention LSTMs: 64 row-tiles x 2 sides, full scan per workgroup
  lstm_phase2_kernel<<<128, 384, 0, stream>>>(XCL, XCR, XRR, pk_wlhh, pk_wrhh, lens, out);
}
